// NgramLanguageModel_15539191677752
// MI455X (gfx1250) — matrix operand order repaired and checked by reading, not compiled, not run on hardware
//
#include <hip/hip_runtime.h>

// N-gram LM forward as a bf16-WMMA tiled GEMM (MI455X / gfx1250, wave32):
//   out[m, n] = sum_k X[m,k] * W[n,k] + bias[n]
//   X[m, seg*V + v] = emb[idx[b, t+seg], v],  m = b*1022 + t
// M=8176, K=16384, N=8192.  Compute-bound => v_wmma_f32_16x16x32_bf16.
//
// Fast path (if async-to-LDS builtins exist and d_ws >= 384MB):
//   1) elementwise fp32->bf16 conversion of W and emb into d_ws
//   2) GEMM stage loop uses GLOBAL_LOAD_ASYNC_TO_LDS_B128 (ASYNCcnt) with
//      3-deep LDS buffering: no VGPR staging, no ds_store, no cvt in loop.
// Fallback: synchronous global->reg->cvt->LDS pipeline (rounds 1-2 kernel).

typedef __attribute__((ext_vector_type(16))) __bf16 v16bf;
typedef __attribute__((ext_vector_type(4)))  __bf16 v4bf;
typedef __attribute__((ext_vector_type(8)))  float  v8f;
typedef __attribute__((ext_vector_type(4)))  float  f32x4;
typedef __attribute__((ext_vector_type(2)))  unsigned int u32x2;
typedef __attribute__((ext_vector_type(4)))  int    i32x4;

#define V_DIM   8192
#define KTOT    16384
#define TN      1022
#define M_ROWS  8176       // 8 * 1022
#define BM      128
#define BN      128
#define BK      64
#define STR     144        // LDS row stride in bytes (BK*2 + 16 pad -> conflict-free b128)
#define STAGES  (KTOT / BK)

#if defined(__has_builtin)
#if __has_builtin(__builtin_amdgcn_global_load_async_to_lds_b128) && \
    __has_builtin(__builtin_amdgcn_s_wait_asynccnt)
#define HAVE_ASYNC 1
#endif
#endif
#ifndef HAVE_ASYNC
#define HAVE_ASYNC 0
#endif

union FragAB { v16bf v; f32x4 f[2]; };
union BF4    { v4bf  b; u32x2 u; };

// ---------------------------------------------------------------------------
// Shared epilogue/compute helpers
// ---------------------------------------------------------------------------
__device__ __forceinline__ void wmma_stage(const unsigned char* Ab,
                                           const unsigned char* Bb,
                                           const int aoff[2], const int boff[4],
                                           v8f acc[2][4]) {
    #pragma unroll
    for (int ks = 0; ks < 2; ++ks) {        // two k=32 WMMA steps per BK=64 stage
        FragAB a[2], bfr[4];
        #pragma unroll
        for (int rt = 0; rt < 2; ++rt) {
            a[rt].f[0] = *(const f32x4*)(Ab + aoff[rt] + ks * 64);
            a[rt].f[1] = *(const f32x4*)(Ab + aoff[rt] + ks * 64 + 16);
        }
        #pragma unroll
        for (int ct = 0; ct < 4; ++ct) {
            bfr[ct].f[0] = *(const f32x4*)(Bb + boff[ct] + ks * 64);
            bfr[ct].f[1] = *(const f32x4*)(Bb + boff[ct] + ks * 64 + 16);
        }
        #pragma unroll
        for (int rt = 0; rt < 2; ++rt)
            #pragma unroll
            for (int ct = 0; ct < 4; ++ct)
                acc[rt][ct] = __builtin_amdgcn_wmma_f32_16x16x32_bf16(
                    false, a[rt].v, false, bfr[ct].v,
                    (short)0, acc[rt][ct], false, false);
    }
}

__device__ __forceinline__ void store_epilogue(v8f acc[2][4], const float* bias,
                                               float* out, int m0, int n0,
                                               int wm, int wn, int l16, int lhi) {
    #pragma unroll
    for (int ct = 0; ct < 4; ++ct) {
        int n = n0 + wn * 64 + ct * 16 + l16;
        float bv = bias[n];
        #pragma unroll
        for (int rt = 0; rt < 2; ++rt) {
            int mbase = m0 + wm * 32 + rt * 16 + lhi * 8;
            #pragma unroll
            for (int v = 0; v < 8; ++v) {
                int m = mbase + v;
                if (m < M_ROWS)
                    out[(size_t)m * V_DIM + n] = acc[rt][ct][v] + bv;
            }
        }
    }
}

// ===========================================================================
// FAST PATH: bf16 operands in d_ws + GLOBAL_LOAD_ASYNC_TO_LDS_B128
// ===========================================================================
#if HAVE_ASYNC

typedef __attribute__((address_space(1))) i32x4 g_i32x4;   // global int4
typedef __attribute__((address_space(3))) i32x4 l_i32x4;   // LDS int4

__device__ __forceinline__ void async_cp16(const void* g, void* l) {
    __builtin_amdgcn_global_load_async_to_lds_b128(
        (g_i32x4*)g, (l_i32x4*)l, 0, 0);
}

__global__ void cvt_f32_bf16(const float* __restrict__ in,
                             unsigned char* __restrict__ outb, int n4) {
    int i = blockIdx.x * blockDim.x + threadIdx.x;
    if (i < n4) {
        f32x4 v = *(const f32x4*)(in + (size_t)i * 4);
        BF4 p; p.b = __builtin_convertvector(v, v4bf);
        *(u32x2*)(outb + (size_t)i * 8) = p.u;
    }
}

__global__ __launch_bounds__(256)
void ngram_wmma_gemm_async(const int* __restrict__ idx,
                           const unsigned char* __restrict__ embB,  // bf16 [V][V]
                           const unsigned char* __restrict__ WB,    // bf16 [V][K]
                           const float* __restrict__ bias,
                           float* __restrict__ out)
{
    __shared__ __align__(16) unsigned char ldsA[3][BM * STR];
    __shared__ __align__(16) unsigned char ldsB[3][BN * STR];
    __shared__ int rowOff[2][BM];

    const int tid = threadIdx.x;
    const int m0  = blockIdx.y * BM;
    const int n0  = blockIdx.x * BN;

    {   // gather byte-offsets (token * V * 2) for both n-gram segments
        int seg = tid >> 7;
        int r   = tid & 127;
        int m   = m0 + r;
        if (m >= M_ROWS) m = M_ROWS - 1;
        int b   = m / TN;
        int t   = m - b * TN;
        rowOff[seg][r] = idx[b * 1024 + t + seg] << 14;
    }
    __syncthreads();

    // Each thread async-copies 4 rows x one 16B chunk per matrix per stage.
    const int rb  = tid >> 3;        // base row (rows rb, rb+32, rb+64, rb+96)
    const int col = tid & 7;         // 16B chunk within the 128B (64 bf16) row
    int rA0[4], rA1[4];
    #pragma unroll
    for (int j = 0; j < 4; ++j) {
        rA0[j] = rowOff[0][rb + j * 32];
        rA1[j] = rowOff[1][rb + j * 32];
    }

    const unsigned char* aptr[4];
    const unsigned char* bptr[4];
    #pragma unroll
    for (int j = 0; j < 4; ++j)
        bptr[j] = WB + (size_t)(n0 + rb + j * 32) * (KTOT * 2) + col * 16;

    const unsigned int ldsChunk = (unsigned)(rb * STR + col * 16);

    auto issueStage = [&](int s, int buf) {
        if (((s * BK) & (V_DIM - 1)) == 0) {          // uniform segment switch
            const int seg = (s * BK) >> 13;
            #pragma unroll
            for (int j = 0; j < 4; ++j)
                aptr[j] = embB + (size_t)(seg ? rA1[j] : rA0[j]) + col * 16;
        }
        #pragma unroll
        for (int j = 0; j < 4; ++j) {
            async_cp16(aptr[j], &ldsA[buf][ldsChunk + j * (32 * STR)]);
            aptr[j] += BK * 2;
            async_cp16(bptr[j], &ldsB[buf][ldsChunk + j * (32 * STR)]);
            bptr[j] += BK * 2;
        }
    };

    // Wave tiling: 8 waves in 4x2 grid, each owns 32(M) x 64(N).
    const int lane = tid & 31;
    const int wv   = tid >> 5;
    const int wm   = wv >> 1;
    const int wn   = wv & 1;
    const int l16  = lane & 15;
    const int lhi  = lane >> 4;

    int aoff[2], boff[4];
    #pragma unroll
    for (int rt = 0; rt < 2; ++rt)
        aoff[rt] = (wm * 32 + rt * 16 + l16) * STR + lhi * 32;
    #pragma unroll
    for (int ct = 0; ct < 4; ++ct)
        boff[ct] = (wn * 64 + ct * 16 + l16) * STR + lhi * 32;

    v8f acc[2][4];
    const v8f vzero = {0.f, 0.f, 0.f, 0.f, 0.f, 0.f, 0.f, 0.f};
    #pragma unroll
    for (int rt = 0; rt < 2; ++rt)
        #pragma unroll
        for (int ct = 0; ct < 4; ++ct)
            acc[rt][ct] = vzero;

    issueStage(0, 0);
    issueStage(1, 1);

    int bufC = 0;
    for (int s = 0; s < STAGES; ++s) {
        if (s + 1 < STAGES) __builtin_amdgcn_s_wait_asynccnt(8);  // stage s done
        else                __builtin_amdgcn_s_wait_asynccnt(0);
        __syncthreads();                       // stage-s data visible block-wide
        if (s + 2 < STAGES) {                  // 3 buffers: target last read s-1
            int bufI = bufC + 2; if (bufI >= 3) bufI -= 3;
            issueStage(s + 2, bufI);
        }
        wmma_stage(ldsA[bufC], ldsB[bufC], aoff, boff, acc);
        bufC = (bufC + 1 == 3) ? 0 : bufC + 1;
    }

    store_epilogue(acc, bias, out, m0, n0, wm, wn, l16, lhi);
}
#endif  // HAVE_ASYNC

// ===========================================================================
// FALLBACK: synchronous global->reg->cvt->LDS (works everywhere)
// ===========================================================================
__global__ __launch_bounds__(256)
void ngram_wmma_gemm(const int* __restrict__ idx,
                     const float* __restrict__ emb,
                     const float* __restrict__ W,
                     const float* __restrict__ bias,
                     float* __restrict__ out)
{
    __shared__ __align__(16) unsigned char ldsA[2][BM * STR];
    __shared__ __align__(16) unsigned char ldsB[2][BN * STR];
    __shared__ int rowOff[2][BM];

    const int tid = threadIdx.x;
    const int m0  = blockIdx.y * BM;
    const int n0  = blockIdx.x * BN;

    {
        int seg = tid >> 7;
        int r   = tid & 127;
        int m   = m0 + r;
        if (m >= M_ROWS) m = M_ROWS - 1;
        int b   = m / TN;
        int t   = m - b * TN;
        rowOff[seg][r] = idx[b * 1024 + t + seg] << 13;
    }
    __syncthreads();

    int rowA0[8], rowA1[8];
    #pragma unroll
    for (int i = 0; i < 8; ++i) {
        rowA0[i] = rowOff[0][(tid >> 4) + i * 16];
        rowA1[i] = rowOff[1][(tid >> 4) + i * 16];
    }

    const int c4 = tid & 15;
    const int rB = tid >> 4;

    const float* aptr[8];
    const float* bptr = W + (size_t)(n0 + rB) * KTOT + c4 * 4;

    f32x4 ar[8], br[8];
    auto gload = [&](int s) {
        if (((s * BK) & (V_DIM - 1)) == 0) {
            const int seg = (s * BK) >> 13;
            #pragma unroll
            for (int i = 0; i < 8; ++i)
                aptr[i] = emb + (size_t)(seg ? rowA1[i] : rowA0[i]) + c4 * 4;
        }
        #pragma unroll
        for (int i = 0; i < 8; ++i) {
            ar[i] = *(const f32x4*)aptr[i];
            aptr[i] += BK;
            br[i] = *(const f32x4*)(bptr + i * (16 * KTOT));
        }
        bptr += BK;
    };

    const unsigned int ldsOff = (unsigned)(rB * STR + c4 * 8);
    auto stlds = [&](int buf) {
        #pragma unroll
        for (int i = 0; i < 8; ++i) {
            BF4 pa, pb;
            pa.b = __builtin_convertvector(ar[i], v4bf);
            pb.b = __builtin_convertvector(br[i], v4bf);
            *(u32x2*)&ldsA[buf][ldsOff + i * (16 * STR)] = pa.u;
            *(u32x2*)&ldsB[buf][ldsOff + i * (16 * STR)] = pb.u;
        }
    };

    const int lane = tid & 31;
    const int wv   = tid >> 5;
    const int wm   = wv >> 1;
    const int wn   = wv & 1;
    const int l16  = lane & 15;
    const int lhi  = lane >> 4;

    int aoff[2], boff[4];
    #pragma unroll
    for (int rt = 0; rt < 2; ++rt)
        aoff[rt] = (wm * 32 + rt * 16 + l16) * STR + lhi * 32;
    #pragma unroll
    for (int ct = 0; ct < 4; ++ct)
        boff[ct] = (wn * 64 + ct * 16 + l16) * STR + lhi * 32;

    v8f acc[2][4];
    const v8f vzero = {0.f, 0.f, 0.f, 0.f, 0.f, 0.f, 0.f, 0.f};
    #pragma unroll
    for (int rt = 0; rt < 2; ++rt)
        #pragma unroll
        for (int ct = 0; ct < 4; ++ct)
            acc[rt][ct] = vzero;

    gload(0);
    stlds(0);
    gload(1);
    __syncthreads();

    for (int s = 0; s < STAGES; ++s) {
        const int cur = s & 1;
        if (s + 1 < STAGES) {
            stlds(cur ^ 1);
            if (s + 2 < STAGES) gload(s + 2);
        }
        wmma_stage(ldsA[cur], ldsB[cur], aoff, boff, acc);
        __syncthreads();
    }

    store_epilogue(acc, bias, out, m0, n0, wm, wn, l16, lhi);
}

// ===========================================================================
extern "C" void kernel_launch(void* const* d_in, const int* in_sizes, int n_in,
                              void* d_out, int out_size, void* d_ws, size_t ws_size,
                              hipStream_t stream) {
    const int*   idx = (const int*)d_in[0];
    const float* emb = (const float*)d_in[1];
    const float* W   = (const float*)d_in[2];
    const float* b   = (const float*)d_in[3];
    float* out = (float*)d_out;

    dim3 grid(V_DIM / BN, (M_ROWS + BM - 1) / BM);   // 64 x 64 tiles
    dim3 block(256);                                 // 8 wave32

#if HAVE_ASYNC
    const size_t wbytes = (size_t)V_DIM * KTOT * 2;  // 256MB bf16 W
    const size_t ebytes = (size_t)V_DIM * V_DIM * 2; // 128MB bf16 emb
    if (ws_size >= wbytes + ebytes) {
        unsigned char* wsW = (unsigned char*)d_ws;
        unsigned char* wsE = wsW + wbytes;
        const int nW4 = (int)((size_t)V_DIM * KTOT / 4);   // 33,554,432
        const int nE4 = (int)((size_t)V_DIM * V_DIM / 4);  // 16,777,216
        cvt_f32_bf16<<<dim3(nW4 / 256), dim3(256), 0, stream>>>(W, wsW, nW4);
        cvt_f32_bf16<<<dim3(nE4 / 256), dim3(256), 0, stream>>>(emb, wsE, nE4);
        ngram_wmma_gemm_async<<<grid, block, 0, stream>>>(idx, wsE, wsW, b, out);
        return;
    }
#endif
    ngram_wmma_gemm<<<grid, block, 0, stream>>>(idx, emb, W, b, out);
}
